// AMADSUpsampler_74603581931897
// MI455X (gfx1250) — matrix operand order repaired and checked by reading, not compiled, not run on hardware
//
#include <hip/hip_runtime.h>
#include <hip/hip_bf16.h>
#include <math.h>

// ---------------------------------------------------------------------------
// AMADS 4x upsampler for MI455X (gfx1250, wave32).
// Heavy op: w_detail conv3x3 64->1024 = implicit GEMM M=32768,N=1024,K=576 on
// v_wmma_f32_16x16x32_bf16. Activation patch staged into LDS via the Tensor
// Data Mover (tensor_load_to_lds) for interior tiles, then re-swizzled once
// per block into WMMA A-fragment order so the inner loop reads one 32B LDS
// fragment per lane (2x ds_load_b128) instead of 16 scalar ds_load_u16.
// ---------------------------------------------------------------------------

typedef __attribute__((ext_vector_type(16))) __bf16        v16bf;
typedef __attribute__((ext_vector_type(8)))  float         v8f;
typedef __attribute__((ext_vector_type(4)))  unsigned int  v4u;
typedef __attribute__((ext_vector_type(8)))  int           v8i;
typedef __attribute__((ext_vector_type(4)))  int           v4i;

#if defined(__has_builtin)
#if __has_builtin(__builtin_amdgcn_tensor_load_to_lds)
#define HAVE_TDM 1
#if __has_include(<hip/amd_detail/amd_gfx1250_TDM.h>)
#define TDM_ARGS6 1   // therock-10.0 headers => 6-arg builtin variant
#endif
#endif
#endif

static constexpr int Bn = 2, Cn = 64, Hn = 128, Wn = 128;
static constexpr int Sn = 4, Pn = 16;
static constexpr int HW = Hn * Wn;                 // 16384
static constexpr int KDET = Cn * 9;                // 576
static constexpr int NDET = Cn * Pn;               // 1024
static constexpr int NPIX = Bn * HW;               // 32768 (GEMM M)
static constexpr int HO = Hn * Sn, WO = Wn * Sn;   // 512
static constexpr int KSTEPS = KDET / 32;           // 18

__device__ __forceinline__ unsigned short f2bf(float f) {
  unsigned int u = __float_as_uint(f);
  unsigned int r = u + 0x7FFFu + ((u >> 16) & 1u);   // RNE
  return (unsigned short)(r >> 16);
}
__device__ __forceinline__ float gelu_exact(float s) {
  return 0.5f * s * (1.0f + erff(s * 0.70710678118654752f));
}

// ---- (1) x -> bf16 ---------------------------------------------------------
__global__ void k_cvt_x(const float* __restrict__ x, unsigned short* __restrict__ xb, int n) {
  int i = blockIdx.x * 256 + threadIdx.x;
  if (i < n) xb[i] = f2bf(x[i]);
}

// ---- (2) w_detail [O,I,3,3] -> bf16 Wb[K=576][N=1024], K = i*9 + ky*3 + kx -
__global__ void k_cvt_wdet(const float* __restrict__ w, unsigned short* __restrict__ wb) {
  int i = blockIdx.x * 256 + threadIdx.x;
  if (i < KDET * NDET) {
    int k = i >> 10;            // / 1024
    int n = i & 1023;
    wb[i] = f2bf(w[n * KDET + k]);
  }
}

// ---- (3) fused geom (dw3x3 -> gelu -> 1x1) + edge gate (3x3 -> mish -> 1x1 -> sigmoid)
__global__ void __launch_bounds__(256) k_geomgate(
    const float* __restrict__ x,
    const float* __restrict__ wdw,   // [64,1,3,3]
    const float* __restrict__ wpw,   // [32,64]
    const float* __restrict__ wg1,   // [16,64,3,3]
    const float* __restrict__ wg2,   // [1,16]
    const float* __restrict__ bg2,   // [1]
    float* __restrict__ geom,        // [B,32,H,W]
    float* __restrict__ alpha)       // [B,H,W]
{
  int t = blockIdx.x * 256 + threadIdx.x;
  if (t >= NPIX) return;
  int b = t >> 14;
  int r = t & (HW - 1);
  int h = r >> 7, w = r & (Wn - 1);

  float g32[32];
  float hs[16];
#pragma unroll
  for (int o = 0; o < 32; ++o) g32[o] = 0.0f;
#pragma unroll
  for (int j = 0; j < 16; ++j) hs[j] = 0.0f;

  for (int c = 0; c < Cn; ++c) {
    const float* xc = x + (size_t)(b * Cn + c) * HW;
    float s = 0.0f;
#pragma unroll
    for (int dy = -1; dy <= 1; ++dy) {
      int hh = h + dy;
      bool okY = (hh >= 0) && (hh < Hn);
#pragma unroll
      for (int dx = -1; dx <= 1; ++dx) {
        int ww = w + dx;
        float xv = (okY && ww >= 0 && ww < Wn) ? xc[hh * Wn + ww] : 0.0f;
        int k = (dy + 1) * 3 + (dx + 1);
        s += xv * wdw[c * 9 + k];
#pragma unroll
        for (int j = 0; j < 16; ++j)
          hs[j] += xv * wg1[(j * Cn + c) * 9 + k];
      }
    }
    float gl = gelu_exact(s);
#pragma unroll
    for (int o = 0; o < 32; ++o) g32[o] += wpw[o * Cn + c] * gl;
  }
#pragma unroll
  for (int o = 0; o < 32; ++o)
    geom[(size_t)(b * 32 + o) * HW + r] = g32[o];

  float acc = bg2[0];
#pragma unroll
  for (int j = 0; j < 16; ++j) {
    float v = hs[j];
    float sp = logf(1.0f + expf(v));   // softplus
    acc += wg2[j] * (v * tanhf(sp));   // mish
  }
  alpha[t] = 1.0f / (1.0f + expf(-acc));
}

// ---- (4) offsets: conv3x3 32->64 -> gelu -> 1x1 64->32, gated by alpha -----
__global__ void __launch_bounds__(256) k_offs(
    const float* __restrict__ geom,  // [B,32,H,W]
    const float* __restrict__ w1,    // [64,32,3,3]
    const float* __restrict__ b1,    // [64]
    const float* __restrict__ w2,    // [32,64]
    const float* __restrict__ alpha, // [B,H,W]
    float* __restrict__ offs)        // [B,32,H,W], ch = 2*p + coord
{
  int t = blockIdx.x * 256 + threadIdx.x;
  if (t >= NPIX) return;
  int b = t >> 14;
  int r = t & (HW - 1);
  int h = r >> 7, w = r & (Wn - 1);

  float t1[64];
#pragma unroll
  for (int o = 0; o < 64; ++o) t1[o] = b1[o];

  for (int c = 0; c < 32; ++c) {
    const float* gc = geom + (size_t)(b * 32 + c) * HW;
#pragma unroll
    for (int dy = -1; dy <= 1; ++dy) {
      int hh = h + dy;
      bool okY = (hh >= 0) && (hh < Hn);
#pragma unroll
      for (int dx = -1; dx <= 1; ++dx) {
        int ww = w + dx;
        float gv = (okY && ww >= 0 && ww < Wn) ? gc[hh * Wn + ww] : 0.0f;
        int k = (dy + 1) * 3 + (dx + 1);
#pragma unroll
        for (int o = 0; o < 64; ++o)
          t1[o] += gv * w1[(o * 32 + c) * 9 + k];
      }
    }
  }
#pragma unroll
  for (int o = 0; o < 64; ++o) t1[o] = gelu_exact(t1[o]);

  float al = alpha[t];
  for (int q = 0; q < 32; ++q) {
    float s = 0.0f;
#pragma unroll
    for (int o = 0; o < 64; ++o) s += w2[q * 64 + o] * t1[o];
    offs[(size_t)(b * 32 + q) * HW + r] = s * al;
  }
}

// ---- (5) grid-sample base into d_out ---------------------------------------
// Normalized-coord round trip collapses: sample coord = clamp(pix_coord, 0, 127).
__global__ void __launch_bounds__(256) k_base(
    const float* __restrict__ x,
    const float* __restrict__ offs,
    float* __restrict__ out)        // [B,64,512,512]
{
  int t = blockIdx.x * 256 + threadIdx.x;
  if (t >= Bn * HO * WO) return;
  int b = t >> 18;
  int r = t & (HO * WO - 1);
  int oy = r >> 9, ox = r & (WO - 1);
  int h = oy >> 2, ri = oy & 3;
  int w = ox >> 2, rj = ox & 3;
  int p = ri * 4 + rj;

  const float ticks[4] = {-0.375f, -0.125f, 0.125f, 0.375f};
  size_t pr = (size_t)h * Wn + w;
  float offx = offs[(size_t)(b * 32 + 2 * p) * HW + pr];
  float offy = offs[(size_t)(b * 32 + 2 * p + 1) * HW + pr];

  float fx = (float)w + ticks[rj] + offx;
  float fy = (float)h + ticks[ri] + offy;
  fx = fminf(fmaxf(fx, 0.0f), (float)(Wn - 1));
  fy = fminf(fmaxf(fy, 0.0f), (float)(Hn - 1));
  float x0f = floorf(fx), y0f = floorf(fy);
  float wx = fx - x0f, wy = fy - y0f;
  int x0 = (int)x0f, y0 = (int)y0f;
  int x1 = min(x0 + 1, Wn - 1), y1 = min(y0 + 1, Hn - 1);
  float w00 = (1.0f - wx) * (1.0f - wy), w01 = wx * (1.0f - wy);
  float w10 = (1.0f - wx) * wy,          w11 = wx * wy;

  const float* xb = x + (size_t)b * Cn * HW;
  size_t i00 = (size_t)y0 * Wn + x0, i01 = (size_t)y0 * Wn + x1;
  size_t i10 = (size_t)y1 * Wn + x0, i11 = (size_t)y1 * Wn + x1;
  float* op = out + ((size_t)(b * Cn) * HO + oy) * WO + ox;
  for (int c = 0; c < Cn; ++c) {
    const float* xc = xb + (size_t)c * HW;
    float v = xc[i00] * w00 + xc[i01] * w01 + xc[i10] * w10 + xc[i11] * w11;
    op[(size_t)c * HO * WO] = v;
  }
}

// ---- (6) detail conv via WMMA bf16, TDM-staged patch, fused epilogue -------
// Block = 16-pixel M-tile; 8 waves x 8 N-tiles = all 1024 output channels.
__global__ void __launch_bounds__(256) k_detail(
    const unsigned short* __restrict__ xb,   // [B,64,H,W] bf16
    const unsigned short* __restrict__ wb,   // [K=576][N=1024] bf16
    const float* __restrict__ bdet,          // [1024]
    const float* __restrict__ alpha,         // [B,H,W]
    float* __restrict__ out)                 // [B,64,512,512] (+=)
{
  // Raw patch in TDM tile order [ci][row][col]: z=64 ch, y=3 rows, x=18 cols.
  __shared__ __align__(32) unsigned short xsraw[Cn * 3 * 18];     // 6912 B
  // A operand pre-swizzled into WMMA fragment order [kstep][lane][elem].
  __shared__ __align__(32) unsigned short aswz[KSTEPS * 32 * 16]; // 18432 B
  __shared__ float als[16];

  const int tid = threadIdx.x;
  const int tile = blockIdx.x;
  const int pbase = tile * 16;
  const int b = pbase >> 14;
  const int rem = pbase & (HW - 1);
  const int h = rem >> 7;
  const int w0 = rem & (Wn - 1);     // multiple of 16 (128 % 16 == 0)

  const bool interior = (h >= 1) && (h <= Hn - 2) && (w0 >= 1) && (w0 + 16 <= Wn - 1);

#if HAVE_TDM
  if (interior) {
    if (tid < 32) {   // one wave issues the DMA; EXEC-0 waves branch around it
      unsigned lds_off = (unsigned)(uintptr_t)(&xsraw[0]);   // flat[31:0] == LDS offset
      unsigned long long gaddr = (unsigned long long)(uintptr_t)
          (xb + ((size_t)(b * Cn) * Hn + (h - 1)) * Wn + (w0 - 1));
      // D# group0: count=1 | lds_addr | global_addr | type=2
      v4u g0 = {1u, lds_off, (unsigned)gaddr,
                (unsigned)((gaddr >> 32) & 0x01FFFFFFull) | 0x80000000u};
      // D# group1: data_size=2B; tensor_dim0=18, tensor_dim1=3;
      //            tile 18 x 3 x 64; dim0_stride=W(128), dim1_stride=HW(16384)
      v8i g1 = {0x00010000, 18 << 16, 3 << 16, 18 << 16,
                3 | (64 << 16), 128, 16384 << 16, 0};
      v4i g2 = {64, 0, 0, 0};   // tensor_dim2 = 64 channels
      v4i g3 = {0, 0, 0, 0};
#if TDM_ARGS6
      v8i g4 = {0, 0, 0, 0, 0, 0, 0, 0};
      __builtin_amdgcn_tensor_load_to_lds(g0, g1, g2, g3, g4, 0);
#else
      __builtin_amdgcn_tensor_load_to_lds(g0, g1, g2, g3, 0);
#endif
      __builtin_amdgcn_s_wait_tensorcnt(0);
    }
  } else
#endif
  {
    // Border tiles (halo precedes tensor start): guarded manual staging.
    for (int idx = tid; idx < Cn * 3 * 18; idx += 256) {
      int ci = idx / 54;
      int rc = idx - ci * 54;
      int row = rc / 18;
      int col = rc - row * 18;
      int hh = h - 1 + row;
      int ww = w0 - 1 + col;
      unsigned short v = 0;
      if (hh >= 0 && hh < Hn && ww >= 0 && ww < Wn)
        v = xb[((size_t)(b * Cn + ci) * Hn + hh) * Wn + ww];
      xsraw[idx] = v;
    }
  }
  if (tid < 16) als[tid] = alpha[pbase + tid];
  __syncthreads();

  // One-time swizzle: raw patch -> A fragments (shared by all 8 waves).
  // aswz[(kt*32+lane)*16 + e] = patch value for WMMA 16-bit A 16x32 layout.
  for (int idx = tid; idx < KSTEPS * 32 * 16; idx += 256) {
    int e  = idx & 15;
    int ls = (idx >> 4) & 31;
    int kt = idx >> 9;
    int laneHi = ls >> 4;
    int m = ls & 15;
    int vv = e >> 1, hb = e & 1;
    int kk = ((vv < 4) ? (vv * 2) : (16 + (vv - 4) * 2)) + laneHi * 8 + hb;
    int K = kt * 32 + kk;
    int ci = K / 9;
    int off = K - ci * 9;
    int dy = off / 3, dxx = off - dy * 3;
    aswz[idx] = xsraw[ci * 54 + dy * 18 + (m + dxx)];
  }
  __syncthreads();

  const int wave = tid >> 5;
  const int lane = tid & 31;
  const int laneHi = lane >> 4;
  const int m = lane & 15;          // A-matrix M row / C-matrix N col

  v8f acc[8];
#pragma unroll
  for (int nt = 0; nt < 8; ++nt)
#pragma unroll
    for (int e = 0; e < 8; ++e) acc[nt][e] = 0.0f;

  for (int kt = 0; kt < KSTEPS; ++kt) {
    // A fragment: one 32B LDS read per lane (2x ds_load_b128).
    v16bf av = *(const v16bf*)&aswz[(kt * 32 + lane) * 16];
    const unsigned short* wrow = wb + (size_t)(kt * 32 + lane) * NDET;
    __builtin_prefetch(wrow + 32 * NDET, 0, 1);       // next K-step's weights
    // Batch all 8 B fragments so they issue as one load clause.
    v16bf bf[8];
#pragma unroll
    for (int nt = 0; nt < 8; ++nt)
#pragma unroll
      for (int s = 0; s < 16; ++s)
        bf[nt][s] = __builtin_bit_cast(__bf16, wb[(size_t)(kt * 32 + ((s < 8) ? s : s + 8) + laneHi * 8) * NDET + (wave * 8 + nt) * 16 + m]);
#pragma unroll
    for (int nt = 0; nt < 8; ++nt)
      acc[nt] = __builtin_amdgcn_wmma_f32_16x16x32_bf16(
          false, av, false, bf[nt], (short)0, acc[nt], false, false);
  }

  // Epilogue: bias, tanh, zero-mean over 16 phases (= 16 N lanes of a tile),
  // scatter through pixel_shuffle, fused alpha-gated accumulate into out.
#pragma unroll
  for (int nt = 0; nt < 8; ++nt) {
    int cch = wave * 8 + nt;                 // output channel c (0..63)
    float bias = bdet[cch * 16 + m];         // o = c*16 + p, p = m
    int ri = m >> 2, rj = m & 3;
#pragma unroll
    for (int i = 0; i < 8; ++i) {
      float v = tanhf(acc[nt][i] + bias);
      float s = v;
      s += __shfl_xor(s, 1, 32);
      s += __shfl_xor(s, 2, 32);
      s += __shfl_xor(s, 4, 32);
      s += __shfl_xor(s, 8, 32);             // sum over the 16 phases
      v -= s * (1.0f / 16.0f);
      int mrow = i + laneHi * 8;             // pixel within tile
      float al = als[mrow];
      int w = w0 + mrow;
      size_t oidx = ((size_t)(b * Cn + cch) * HO + (h * 4 + ri)) * WO + (w * 4 + rj);
      out[oidx] += al * 0.25f * v;
    }
  }
}

// ---------------------------------------------------------------------------
extern "C" void kernel_launch(void* const* d_in, const int* in_sizes, int n_in,
                              void* d_out, int out_size, void* d_ws, size_t ws_size,
                              hipStream_t stream) {
  const float* x     = (const float*)d_in[0];
  const float* wdw   = (const float*)d_in[1];
  const float* wpw   = (const float*)d_in[2];
  const float* woff1 = (const float*)d_in[3];
  const float* boff1 = (const float*)d_in[4];
  const float* woff2 = (const float*)d_in[5];
  const float* wg1   = (const float*)d_in[6];
  const float* wg2   = (const float*)d_in[7];
  const float* bg2   = (const float*)d_in[8];
  const float* wdet  = (const float*)d_in[9];
  const float* bdet  = (const float*)d_in[10];
  float* out = (float*)d_out;

  // Workspace carve-up (256B aligned): ~13.9 MB total.
  char* ws = (char*)d_ws;
  size_t o = 0;
  auto take = [&](size_t bytes) { char* p = ws + o; o += (bytes + 255) & ~(size_t)255; return p; };
  unsigned short* xbf   = (unsigned short*)take((size_t)Bn * Cn * HW * 2);
  unsigned short* wbf   = (unsigned short*)take((size_t)KDET * NDET * 2);
  float*          geom  = (float*)take((size_t)Bn * 32 * HW * 4);
  float*          alpha = (float*)take((size_t)Bn * HW * 4);
  float*          offs  = (float*)take((size_t)Bn * 32 * HW * 4);
  (void)ws_size; (void)n_in; (void)in_sizes; (void)out_size;

  int nx = Bn * Cn * HW;
  k_cvt_x<<<(nx + 255) / 256, 256, 0, stream>>>(x, xbf, nx);
  int nw = KDET * NDET;
  k_cvt_wdet<<<(nw + 255) / 256, 256, 0, stream>>>(wdet, wbf);

  k_geomgate<<<(NPIX + 255) / 256, 256, 0, stream>>>(x, wdw, wpw, wg1, wg2, bg2, geom, alpha);
  k_offs<<<(NPIX + 255) / 256, 256, 0, stream>>>(geom, woff1, boff1, woff2, alpha, offs);

  int nout = Bn * HO * WO;
  k_base<<<(nout + 255) / 256, 256, 0, stream>>>(x, offs, out);

  k_detail<<<NPIX / 16, 256, 0, stream>>>(xbf, wbf, bdet, alpha, out);
}
